// CompleteNetwork_20547123544611
// MI455X (gfx1250) — matrix operand order repaired and checked by reading, not compiled, not run on hardware
//
#include <hip/hip_runtime.h>

// ---------------------------------------------------------------------------
// Types for CDNA5 WMMA
// ---------------------------------------------------------------------------
typedef __attribute__((ext_vector_type(16))) __bf16 v16bf;
typedef __attribute__((ext_vector_type(8)))  float  v8f;

// Async global->LDS path (gfx1250); guarded so compile never breaks.
#if defined(__has_builtin)
# if __has_builtin(__builtin_amdgcn_global_load_async_to_lds_b64) && \
     __has_builtin(__builtin_amdgcn_s_wait_asynccnt)
#  define USE_ASYNC_LDS 1
# endif
#endif
#ifndef USE_ASYNC_LDS
# define USE_ASYNC_LDS 0
#endif

#if USE_ASYNC_LDS
// builtin prototype (from hipcc diagnostic): first param is a pointer to
// 'int __vector(2)' in the global/device AS; second is the LDS-side pointer.
typedef int v2i __attribute__((vector_size(2 * sizeof(int))));
typedef __attribute__((address_space(1))) v2i* gas_b64_ptr;
typedef __attribute__((address_space(3))) v2i* las_b64_ptr;
#endif

__device__ __forceinline__ unsigned short f32_to_bf16(float f) {
    unsigned int u = __builtin_bit_cast(unsigned int, f);
    unsigned int r = u + 0x7FFFu + ((u >> 16) & 1u);   // round-to-nearest-even
    return (unsigned short)(r >> 16);
}
__device__ __forceinline__ float bf16_to_f32(unsigned short h) {
    unsigned int u = ((unsigned int)h) << 16;
    return __builtin_bit_cast(float, u);
}

#define N_ATOMS 6000
#define N_PAD   6016     // multiple of 64 for the 64-row GEMM tiles
#define F1 128
#define F2 256
#define F3 512
#define NRES 300
#define NRES_PAD 320     // multiple of 64
#define KNB 10

// ---------------------------------------------------------------------------
// Category extraction: argmax of one-hot rows -> int index per atom
// ---------------------------------------------------------------------------
__global__ void cat_kernel(const float* __restrict__ X, int* __restrict__ out,
                           int n, int c) {
    int i = blockIdx.x * blockDim.x + threadIdx.x;
    if (i >= n) return;
    const float* row = X + (size_t)i * c;
    int best = 0; float bv = row[0];
    for (int j = 1; j < c; ++j) { float v = row[j]; if (v > bv) { bv = v; best = j; } }
    out[i] = best;
}

// ---------------------------------------------------------------------------
// Weight transpose + fp32 -> bf16 : W[K x F] -> Wt[F x K]
// ---------------------------------------------------------------------------
__global__ void transpose_bf16_kernel(const float* __restrict__ W,
                                      unsigned short* __restrict__ Wt,
                                      int K, int F) {
    int tid = blockIdx.x * blockDim.x + threadIdx.x;
    if (tid >= K * F) return;
    int k = tid / F, f = tid % F;
    Wt[(size_t)f * K + k] = f32_to_bf16(W[tid]);
}

// ---------------------------------------------------------------------------
// Layer 1: one-hot matmuls are row gathers. z1 = relu(Wv[a] + Wr[r]
//           + mean_valid(Wsr1[a_sn]) + mean_valid(Wdr1[a_dn]))  -> bf16
// Writes n_pad rows; rows >= n_valid are zero-filled padding.
// ---------------------------------------------------------------------------
__global__ void layer1_kernel(const int* __restrict__ ta, const int* __restrict__ tr,
                              const float* __restrict__ Wv, const float* __restrict__ Wr,
                              const float* __restrict__ Wsr1, const float* __restrict__ Wdr1,
                              const int* __restrict__ sn, const int* __restrict__ dn,
                              unsigned short* __restrict__ z1, int n_valid, int n_pad) {
    int tid = blockIdx.x * blockDim.x + threadIdx.x;
    if (tid >= n_pad * F1) return;
    int a = tid >> 7, f = tid & (F1 - 1);
    if (a >= n_valid) { z1[tid] = 0; return; }
    float z = Wv[ta[a] * F1 + f] + Wr[tr[a] * F1 + f];
    float s = 0.f; int c = 0;
    for (int k = 0; k < KNB; ++k) {
        int idx = sn[a * KNB + k];
        if (idx > -1) { s += Wsr1[ta[idx] * F1 + f]; ++c; }
    }
    z += s / (float)(c > 0 ? c : 1);
    s = 0.f; c = 0;
    for (int k = 0; k < KNB; ++k) {
        int idx = dn[a * KNB + k];
        if (idx > -1) { s += Wdr1[ta[idx] * F1 + f]; ++c; }
    }
    z += s / (float)(c > 0 ? c : 1);
    z = fmaxf(z, 0.f);
    z1[tid] = f32_to_bf16(z);
}

// ---------------------------------------------------------------------------
// WMMA GEMM: Y[M x F] = X[M x K](bf16) @ Wt[F x K](bf16, pre-transposed), f32 out
// block = 128 threads (4 wave32). Block tile: 64(M) x 64(N).
// Wave w owns N-columns [bx*64 + w*16) and 4 M sub-tiles -> 4 accumulators,
// reusing one B fragment across 4 WMMAs per K-step (4x matrix-op density).
// A tile (64x32 bf16, 4KB) staged in LDS via async global->LDS when available.
// Requires M % 64 == 0, K % 32 == 0, F % 64 == 0.
// ---------------------------------------------------------------------------
__global__ void __launch_bounds__(128)
wmma_gemm_bf16(const unsigned short* __restrict__ X,
               const unsigned short* __restrict__ Wt,
               float* __restrict__ Y, int K, int F) {
    __shared__ __align__(16) unsigned short sA[64][32];

    const int t    = threadIdx.x;
    const int wave = t >> 5;
    const int lane = t & 31;
    const int half = lane >> 4;        // lanes 0-15 vs 16-31
    const int ln   = lane & 15;
    const int tile_m = blockIdx.y << 6;
    const int ncol   = (blockIdx.x << 6) + (wave << 4) + ln;

    const unsigned short* wrow = Wt + (size_t)ncol * K + 16 * half;

    v8f acc[4] = {};

    for (int k0 = 0; k0 < K; k0 += 32) {
        __syncthreads();   // previous iteration's LDS reads complete
        // stage 64x32 bf16 A tile = 512 8-byte chunks, 4 chunks per thread
#if USE_ASYNC_LDS
#pragma unroll
        for (int i = 0; i < 4; ++i) {
            int c    = t + (i << 7);
            int row  = c >> 3;
            int off8 = (c & 7) << 2;   // offset in ushort units (8B chunks)
            v2i* gp = (v2i*)(X + (size_t)(tile_m + row) * K + k0 + off8);
            v2i* lp = (v2i*)(&sA[row][off8]);
            __builtin_amdgcn_global_load_async_to_lds_b64(
                (gas_b64_ptr)gp, (las_b64_ptr)lp, 0, 0);
        }
        __builtin_amdgcn_s_wait_asynccnt(0);
#else
#pragma unroll
        for (int i = 0; i < 4; ++i) {
            int c    = t + (i << 7);
            int row  = c >> 3;
            int off8 = (c & 7) << 2;
            *(uint2*)&sA[row][off8] =
                *(const uint2*)(X + (size_t)(tile_m + row) * K + k0 + off8);
        }
#endif
        __syncthreads();   // A tile visible to all waves

        // B fragment (32x16): N = ncol, K = k0 + 16*half .. +15 contiguous
        // in transposed weights -> two 16B global reads, held across 4 WMMAs.
        union { uint4 q[2]; v16bf v; } b;
        const unsigned short* wp = wrow + k0;
        b.q[0] = ((const uint4*)wp)[0];
        b.q[1] = ((const uint4*)wp)[1];
        if (k0 + 32 < K) __builtin_prefetch(wp + 32, 0, 3);  // next K-slice of W

#pragma unroll
        for (int s = 0; s < 4; ++s) {
            union { uint4 q[2]; v16bf v; } a;
            // A fragment (16-bit 16x32): lane ln holds M=ln; VGPR0-7 hold
            // K=16*half..+15 contiguous (matches B)
            const unsigned short* ap = &sA[(s << 4) + ln][0];
            a.q[0] = *(const uint4*)(ap + 16 * half);
            a.q[1] = *(const uint4*)(ap + 16 * half + 8);
            acc[s] = __builtin_amdgcn_wmma_f32_16x16x32_bf16(
                /*neg_a=*/false, a.v, /*neg_b=*/false, b.v,
                /*c_mod=*/(short)0, acc[s], /*reuse_a=*/false, /*reuse_b=*/false);
        }
    }

    // C/D layout: VGPR r -> M = r + 8*half, N = ln
#pragma unroll
    for (int s = 0; s < 4; ++s) {
        float* yrow = Y + (size_t)(tile_m + (s << 4) + 8 * half) * F + ncol;
#pragma unroll
        for (int r = 0; r < 8; ++r) yrow[(size_t)r * F] = acc[s][r];
    }
}

// ---------------------------------------------------------------------------
// Combine: z_next = relu(Ysv + mean_valid(Ysr[sn]) + mean_valid(Ydr[dn])) -> bf16
// Writes n_pad rows; rows >= n_valid are zero-filled padding.
// ---------------------------------------------------------------------------
__global__ void combine_kernel(const float* __restrict__ Ysv, const float* __restrict__ Ysr,
                               const float* __restrict__ Ydr,
                               const int* __restrict__ sn, const int* __restrict__ dn,
                               unsigned short* __restrict__ zout,
                               int n_valid, int n_pad, int F) {
    int tid = blockIdx.x * blockDim.x + threadIdx.x;
    if (tid >= n_pad * F) return;
    int a = tid / F, f = tid % F;
    if (a >= n_valid) { zout[tid] = 0; return; }
    float z = Ysv[(size_t)a * F + f];
    float s = 0.f; int c = 0;
    for (int k = 0; k < KNB; ++k) {
        int idx = sn[a * KNB + k];
        if (idx > -1) { s += Ysr[(size_t)idx * F + f]; ++c; }
    }
    z += s / (float)(c > 0 ? c : 1);
    s = 0.f; c = 0;
    for (int k = 0; k < KNB; ++k) {
        int idx = dn[a * KNB + k];
        if (idx > -1) { s += Ydr[(size_t)idx * F + f]; ++c; }
    }
    z += s / (float)(c > 0 ? c : 1);
    zout[tid] = f32_to_bf16(fmaxf(z, 0.f));
}

// ---------------------------------------------------------------------------
// Residue pooling: mean over 20 consecutive atoms; rows >= NRES zero padding
// ---------------------------------------------------------------------------
__global__ void pool_kernel(const unsigned short* __restrict__ z3,
                            unsigned short* __restrict__ r) {
    int tid = blockIdx.x * blockDim.x + threadIdx.x;
    if (tid >= NRES_PAD * F3) return;
    int res = tid >> 9, f = tid & (F3 - 1);
    float s = 0.f;
    if (res < NRES) {
        for (int a = 0; a < 20; ++a) s += bf16_to_f32(z3[(size_t)(res * 20 + a) * F3 + f]);
        s *= (1.f / 20.f);
    }
    r[tid] = f32_to_bf16(s);
}

// ---------------------------------------------------------------------------
// FC head factorization (fully linear): w23 = fcw2 @ fcw3 ; c = <b1,w23>+<b2,fcw3>+b3
// ---------------------------------------------------------------------------
__global__ void w23_kernel(const float* __restrict__ fcw2, const float* __restrict__ fcw3,
                           const float* __restrict__ fcb1, const float* __restrict__ fcb2,
                           const float* __restrict__ fcb3, float* __restrict__ w23c) {
    int i = threadIdx.x;  // 128 threads, 1 block
    float s = 0.f;
    for (int j = 0; j < 64; ++j) s += fcw2[i * 64 + j] * fcw3[j];
    w23c[i] = s;
    __syncthreads();
    if (i == 0) {
        float c = fcb3[0];
        for (int k = 0; k < 128; ++k) c += fcb1[k] * w23c[k];
        for (int j = 0; j < 64; ++j)  c += fcb2[j] * fcw3[j];
        w23c[128] = c;
    }
}

__global__ void uv_kernel(const float* __restrict__ U, const float* __restrict__ V,
                          const float* __restrict__ w23c,
                          float* __restrict__ u, float* __restrict__ v) {
    int tid = blockIdx.x * blockDim.x + threadIdx.x;
    if (tid < NRES_PAD) {
        float s = 0.f;
        for (int k = 0; k < 128; ++k) s += U[(size_t)tid * 128 + k] * w23c[k];
        u[tid] = s;
    } else if (tid < 2 * NRES_PAD) {
        int j = tid - NRES_PAD;
        float s = 0.f;
        for (int k = 0; k < 128; ++k) s += V[(size_t)j * 128 + k] * w23c[k];
        v[j] = s;
    }
}

// h_ij = u_i + v_j + c ; log_softmax over the size-1 axis == h - logsumexp(h) == 0
__global__ void final_kernel(const float* __restrict__ u, const float* __restrict__ v,
                             const float* __restrict__ w23c, float* __restrict__ out) {
    int p = blockIdx.x * blockDim.x + threadIdx.x;
    if (p >= NRES * NRES) return;
    int i = p / NRES, j = p % NRES;
    float h = u[i] + v[j] + w23c[128];
    float lse = h;                 // logsumexp over a single element
    out[p] = h - lse;
}

// ---------------------------------------------------------------------------
// Host orchestration
// ---------------------------------------------------------------------------
extern "C" void kernel_launch(void* const* d_in, const int* in_sizes, int n_in,
                              void* d_out, int out_size, void* d_ws, size_t ws_size,
                              hipStream_t stream) {
    (void)in_sizes; (void)n_in; (void)out_size; (void)ws_size;

    // inputs (setup_inputs order)
    const float* atoms[2]    = { (const float*)d_in[0], (const float*)d_in[2] };
    const float* residues[2] = { (const float*)d_in[1], (const float*)d_in[3] };
    const int*   sn[2]       = { (const int*)d_in[4], (const int*)d_in[6] };
    const int*   dn[2]       = { (const int*)d_in[5], (const int*)d_in[7] };
    const float* Wv   = (const float*)d_in[10];
    const float* Wr   = (const float*)d_in[11];
    const float* Wsr1 = (const float*)d_in[12];
    const float* Wdr1 = (const float*)d_in[13];
    const float* Wsv2 = (const float*)d_in[14];
    const float* Wsr2 = (const float*)d_in[15];
    const float* Wdr2 = (const float*)d_in[16];
    const float* Wsv3 = (const float*)d_in[17];
    const float* Wsr3 = (const float*)d_in[18];
    const float* Wdr3 = (const float*)d_in[19];
    const float* fcw1 = (const float*)d_in[20];
    const float* fcb1 = (const float*)d_in[21];
    const float* fcw2 = (const float*)d_in[22];
    const float* fcb2 = (const float*)d_in[23];
    const float* fcw3 = (const float*)d_in[24];
    const float* fcb3 = (const float*)d_in[25];

    // workspace carve-up (256B aligned slabs)
    char* ws = (char*)d_ws;
    size_t off = 0;
    auto take = [&](size_t bytes) -> void* {
        void* p = ws + off;
        off += (bytes + 255) & ~(size_t)255;
        return p;
    };
    int* ta = (int*)take(N_ATOMS * 4);
    int* tr = (int*)take(N_ATOMS * 4);
    unsigned short* Wsv2t = (unsigned short*)take((size_t)F2 * F1 * 2);
    unsigned short* Wsr2t = (unsigned short*)take((size_t)F2 * F1 * 2);
    unsigned short* Wdr2t = (unsigned short*)take((size_t)F2 * F1 * 2);
    unsigned short* Wsv3t = (unsigned short*)take((size_t)F3 * F2 * 2);
    unsigned short* Wsr3t = (unsigned short*)take((size_t)F3 * F2 * 2);
    unsigned short* Wdr3t = (unsigned short*)take((size_t)F3 * F2 * 2);
    unsigned short* F1at  = (unsigned short*)take((size_t)128 * F3 * 2);
    unsigned short* F1bt  = (unsigned short*)take((size_t)128 * F3 * 2);
    unsigned short* z1 = (unsigned short*)take((size_t)N_PAD * F1 * 2);
    unsigned short* z2 = (unsigned short*)take((size_t)N_PAD * F2 * 2);
    unsigned short* z3 = (unsigned short*)take((size_t)N_PAD * F3 * 2);
    float* Ysv = (float*)take((size_t)N_PAD * F3 * 4);
    float* Ysr = (float*)take((size_t)N_PAD * F3 * 4);
    float* Ydr = (float*)take((size_t)N_PAD * F3 * 4);
    unsigned short* rb[2] = { (unsigned short*)take((size_t)NRES_PAD * F3 * 2),
                              (unsigned short*)take((size_t)NRES_PAD * F3 * 2) };
    float* U    = (float*)take((size_t)NRES_PAD * 128 * 4);
    float* V    = (float*)take((size_t)NRES_PAD * 128 * 4);
    float* w23c = (float*)take(129 * 4);
    float* uvec = (float*)take(NRES_PAD * 4);
    float* vvec = (float*)take(NRES_PAD * 4);

    const int TB = 256;
    auto cdiv = [](int a, int b) { return (a + b - 1) / b; };

    // pre-transpose / convert weights to bf16 (B-fragment friendly layout)
    transpose_bf16_kernel<<<cdiv(F1 * F2, TB), TB, 0, stream>>>(Wsv2, Wsv2t, F1, F2);
    transpose_bf16_kernel<<<cdiv(F1 * F2, TB), TB, 0, stream>>>(Wsr2, Wsr2t, F1, F2);
    transpose_bf16_kernel<<<cdiv(F1 * F2, TB), TB, 0, stream>>>(Wdr2, Wdr2t, F1, F2);
    transpose_bf16_kernel<<<cdiv(F2 * F3, TB), TB, 0, stream>>>(Wsv3, Wsv3t, F2, F3);
    transpose_bf16_kernel<<<cdiv(F2 * F3, TB), TB, 0, stream>>>(Wsr3, Wsr3t, F2, F3);
    transpose_bf16_kernel<<<cdiv(F2 * F3, TB), TB, 0, stream>>>(Wdr3, Wdr3t, F2, F3);
    transpose_bf16_kernel<<<cdiv(F3 * 128, TB), TB, 0, stream>>>(fcw1,                    F1at, F3, 128);
    transpose_bf16_kernel<<<cdiv(F3 * 128, TB), TB, 0, stream>>>(fcw1 + (size_t)F3 * 128, F1bt, F3, 128);

    for (int p = 0; p < 2; ++p) {
        cat_kernel<<<cdiv(N_ATOMS, TB), TB, 0, stream>>>(atoms[p],    ta, N_ATOMS, 12);
        cat_kernel<<<cdiv(N_ATOMS, TB), TB, 0, stream>>>(residues[p], tr, N_ATOMS, 23);
        layer1_kernel<<<cdiv(N_PAD * F1, TB), TB, 0, stream>>>(
            ta, tr, Wv, Wr, Wsr1, Wdr1, sn[p], dn[p], z1, N_ATOMS, N_PAD);

        // layer 2: three 6016x128x256 WMMA GEMMs
        dim3 g2(F2 / 64, N_PAD / 64);
        wmma_gemm_bf16<<<g2, 128, 0, stream>>>(z1, Wsv2t, Ysv, F1, F2);
        wmma_gemm_bf16<<<g2, 128, 0, stream>>>(z1, Wsr2t, Ysr, F1, F2);
        wmma_gemm_bf16<<<g2, 128, 0, stream>>>(z1, Wdr2t, Ydr, F1, F2);
        combine_kernel<<<cdiv(N_PAD * F2, TB), TB, 0, stream>>>(
            Ysv, Ysr, Ydr, sn[p], dn[p], z2, N_ATOMS, N_PAD, F2);

        // layer 3: three 6016x256x512 WMMA GEMMs
        dim3 g3(F3 / 64, N_PAD / 64);
        wmma_gemm_bf16<<<g3, 128, 0, stream>>>(z2, Wsv3t, Ysv, F2, F3);
        wmma_gemm_bf16<<<g3, 128, 0, stream>>>(z2, Wsr3t, Ysr, F2, F3);
        wmma_gemm_bf16<<<g3, 128, 0, stream>>>(z2, Wdr3t, Ydr, F2, F3);
        combine_kernel<<<cdiv(N_PAD * F3, TB), TB, 0, stream>>>(
            Ysv, Ysr, Ydr, sn[p], dn[p], z3, N_ATOMS, N_PAD, F3);

        pool_kernel<<<cdiv(NRES_PAD * F3, TB), TB, 0, stream>>>(z3, rb[p]);
    }

    // FC head (factored, fully linear): U = r1 @ fcw1_top, V = r2 @ fcw1_bot
    dim3 gf(128 / 64, NRES_PAD / 64);
    wmma_gemm_bf16<<<gf, 128, 0, stream>>>(rb[0], F1at, U, F3, 128);
    wmma_gemm_bf16<<<gf, 128, 0, stream>>>(rb[1], F1bt, V, F3, 128);
    w23_kernel<<<1, 128, 0, stream>>>(fcw2, fcw3, fcb1, fcb2, fcb3, w23c);
    uv_kernel<<<cdiv(2 * NRES_PAD, TB), TB, 0, stream>>>(U, V, w23c, uvec, vvec);
    final_kernel<<<cdiv(NRES * NRES, TB), TB, 0, stream>>>(uvec, vvec, w23c, (float*)d_out);
}
